// CDGM_69406671503635
// MI455X (gfx1250) — matrix operand order repaired and checked by reading, not compiled, not run on hardware
//
#include <hip/hip_runtime.h>
#include <hip/hip_bf16.h>
#include <math.h>

typedef __attribute__((ext_vector_type(16))) __bf16 v16bf;
typedef __attribute__((ext_vector_type(8)))  __bf16 v8bf;
typedef __attribute__((ext_vector_type(8)))  float  v8f;
typedef __attribute__((ext_vector_type(4)))  unsigned v4u;
typedef __attribute__((ext_vector_type(8)))  int      v8i;
typedef __attribute__((ext_vector_type(4)))  int      v4i;

#define N_TOT 8192
#define EPS32 1.1920929e-07f

static __device__ __forceinline__ v8f wmma_bf16(v16bf a, v16bf b, v8f c) {
    return __builtin_amdgcn_wmma_f32_16x16x32_bf16(false, a, false, b, (short)0, c, false, false);
}

// ---------------------------------------------------------------------------
// TDM 2-D tile load: global (bf16, row stride stride0 elems) -> LDS.
// Builds D# group0/group1 per CDNA5 ISA 8.3/8.4 and issues TENSOR_LOAD_TO_LDS
// (6-arg toolchain variant: g0, g1, g2, g3, extra group, cpol).
// dims in 2-byte elements. Tracked by TENSORcnt.
// ---------------------------------------------------------------------------
static __device__ __forceinline__ void tdm_load_2d(unsigned lds_off, const void* gptr,
                                                   unsigned tile_d0, unsigned tile_d1,
                                                   unsigned stride0)
{
    unsigned long long ga = (unsigned long long)(uintptr_t)gptr;
    v4u g0;
    g0[0] = 1u;                                            // count=1, user descriptor
    g0[1] = lds_off;                                       // lds_addr [63:32]
    g0[2] = (unsigned)(ga & 0xFFFFFFFFu);                  // global_addr lo
    g0[3] = (unsigned)((ga >> 32) & 0x1FFFFFFu) | (2u << 30); // addr hi + type=2
    v8i g1;
    g1[0] = (int)(1u << 16);                               // wg_mask=0, data_size=2B
    g1[1] = (int)((tile_d0 & 0xFFFFu) << 16);              // tensor_dim0 lo16 (== tile)
    g1[2] = (int)((tile_d1 & 0xFFFFu) << 16);              // td0 hi=0 | tensor_dim1 lo16
    g1[3] = (int)((tile_d0 & 0xFFFFu) << 16);              // td1 hi=0 | tile_dim0
    g1[4] = (int)(tile_d1 & 0xFFFFu);                      // tile_dim1 | tile_dim2=0
    g1[5] = (int)stride0;                                  // tensor_dim0_stride lo32
    g1[6] = 0;                                             // stride0 hi16 | dim1_stride lo16
    g1[7] = 0;
    v4i z4 = {0, 0, 0, 0};
    v8i z8 = {0, 0, 0, 0, 0, 0, 0, 0};
    __builtin_amdgcn_tensor_load_to_lds(g0, g1, z4, z4, z8, 0);
}

// ---------------------------------------------------------------------------
// Prep: gl = relu(x@Wgl+bgl) -> bf16 [N,FO];  sq[n] = sum(gl^2) fp32;
//       h  = x@Wgnn+bgnn     -> bf16 transposed hT [FO,N]
// ---------------------------------------------------------------------------
template<int FI, int FO>
__global__ void prep_layer(const float* __restrict__ x,
                           const float* __restrict__ Wgl,  const float* __restrict__ bgl,
                           const float* __restrict__ Wgnn, const float* __restrict__ bgnn,
                           __bf16* __restrict__ gl, __bf16* __restrict__ hT,
                           float* __restrict__ sq)
{
    const int wave = threadIdx.x >> 5, lane = threadIdx.x & 31;
    const int row  = blockIdx.x * 8 + wave;
    const float* xr = x + (size_t)row * FI;
    float ss = 0.0f;
#pragma unroll
    for (int u = 0; u < FO / 32; ++u) {
        const int f = lane + 32 * u;
        float ag = 0.0f, ah = 0.0f;
        for (int k = 0; k < FI; ++k) {
            const float xv = xr[k];
            ag = fmaf(xv, Wgl [k * FO + f], ag);
            ah = fmaf(xv, Wgnn[k * FO + f], ah);
        }
        const float g = fmaxf(ag + bgl[f], 0.0f);
        const float h = ah + bgnn[f];
        gl[(size_t)row * FO + f]    = (__bf16)g;
        hT[(size_t)f * N_TOT + row] = (__bf16)h;
        ss += g * g;
    }
#pragma unroll
    for (int off = 16; off > 0; off >>= 1) ss += __shfl_xor(ss, off, 32);
    if (lane == 0) sq[row] = ss;
}

// ---------------------------------------------------------------------------
// Fused adjacency layer (flash-attention style), TDM-staged operands:
//   wave0 DMAs gl_j [32][F] and hT-slice [F][32] tiles into double-buffered
//   LDS via TENSOR_LOAD_TO_LDS (s_wait_tensorcnt + barrier handoff);
//   all 8 waves compute S = gl_i @ gl_j^T, adj = sigmoid(...), then
//   acc += adj @ h and deg += adj @ ones via bf16 WMMA, fp32 accumulation.
// ---------------------------------------------------------------------------
template<int F, bool RELU>
__global__ void __launch_bounds__(256)
fused_adj_layer(const __bf16* __restrict__ gl,
                const __bf16* __restrict__ hT,
                const float*  __restrict__ sq,
                const float*  __restrict__ tempp,
                const float*  __restrict__ thetap,
                float* __restrict__ xout)
{
    constexpr int KC = F / 32;   // K chunks for S
    constexpr int FC = F / 16;   // output column blocks
    __shared__ __align__(64) __bf16 ldsA[8][16][32];    // per-wave 16x32 adj tile
    __shared__ __align__(64) __bf16 ldsG[8][16][F];     // per-wave gl_i stripe
    __shared__ __align__(64) __bf16 stageG[2][32][F];   // TDM: gl_j tiles
    __shared__ __align__(64) __bf16 stageH[2][F][32];   // TDM: hT tiles

    const int wave = threadIdx.x >> 5, lane = threadIdx.x & 31;
    const int n16 = lane & 15, hi = lane >> 4;
    const int ibase = blockIdx.x * 128 + wave * 16;
    const float tv = tempp[0], th = thetap[0];
    const float c1 = 1.0f + tv, c0 = 5.0f + th;

    // Park this stripe's gl rows in per-wave LDS (row-major [16][F]).
#pragma unroll
    for (int c = 0; c < F / 2; c += 8) {
        *(v8bf*)&ldsG[wave][n16][hi * (F / 2) + c] =
            *(const v8bf*)(gl + (size_t)(ibase + n16) * F + hi * (F / 2) + c);
    }
    asm volatile("s_wait_dscnt 0" ::: "memory");

    // Prologue: DMA first j-tile into buffer 0.
    if (wave == 0) {
        tdm_load_2d((unsigned)(uintptr_t)&stageG[0][0][0], gl, F, 32, F);
        tdm_load_2d((unsigned)(uintptr_t)&stageH[0][0][0], hT, 32, F, N_TOT);
        __builtin_amdgcn_s_wait_tensorcnt(0);
    }
    __syncthreads();

    float si[8];
#pragma unroll
    for (int v = 0; v < 8; ++v) si[v] = sq[ibase + v + 8 * hi];

    v8f zero = {};
    v8f acc[FC];
#pragma unroll
    for (int fc = 0; fc < FC; ++fc) acc[fc] = zero;
    v8f deg = zero;
    v16bf ones;
#pragma unroll
    for (int e = 0; e < 16; ++e) ones[e] = (__bf16)1.0f;

    for (int j = 0; j < N_TOT; j += 32) {
        const int b = (j >> 5) & 1;
        // Kick off DMA of the next tile pair into the other buffer.
        if (wave == 0 && j + 32 < N_TOT) {
            tdm_load_2d((unsigned)(uintptr_t)&stageG[b ^ 1][0][0],
                        gl + (size_t)(j + 32) * F, F, 32, F);
            tdm_load_2d((unsigned)(uintptr_t)&stageH[b ^ 1][0][0],
                        hT + (j + 32), 32, F, N_TOT);
        }

        // S tiles for both 16-column halves; A loaded once per K-chunk.
        v8f s0 = zero, s1 = zero;
#pragma unroll
        for (int kc = 0; kc < KC; ++kc) {
            // A 16x32 bf16: lane<16: M=lane, K 0..15; lane>=16: M=lane-16, K 16..31
            const __bf16* pa = &ldsG[wave][n16][kc * 32 + hi * 16];
            v8bf alo = *(const v8bf*)pa;
            v8bf ahi = *(const v8bf*)(pa + 8);
            v16bf Ag = __builtin_shufflevector(alo, ahi,
                           0,1,2,3,4,5,6,7,8,9,10,11,12,13,14,15);
            // B 32x16 bf16: lane<16 holds K=0..15 of col n16; lane>=16 K=16..31
            v16bf B0 = *(const v16bf*)&stageG[b][n16][kc * 32 + hi * 16];
            v16bf B1 = *(const v16bf*)&stageG[b][16 + n16][kc * 32 + hi * 16];
            s0 = wmma_bf16(Ag, B0, s0);
            s1 = wmma_bf16(Ag, B1, s1);
        }
#pragma unroll
        for (int t = 0; t < 2; ++t) {
            const v8f& s = t ? s1 : s0;
            const float sj = sq[j + t * 16 + n16];
#pragma unroll
            for (int v = 0; v < 8; ++v) {
                float d    = fmaxf(si[v] + sj - 2.0f * s[v], 0.0f);
                float msk  = (d != 0.0f) ? 1.0f : 0.0f;
                float dist = -__builtin_amdgcn_sqrtf(d + EPS32) * msk;   // v_sqrt_f32
                float z    = c1 * dist + c0;
                float a    = __builtin_amdgcn_rcpf(1.0f + __expf(-z));   // v_rcp_f32
                ldsA[wave][v + 8 * hi][t * 16 + n16] = (__bf16)a;        // C layout -> [m][k]
            }
        }
        // Intra-wave LDS fence: adj stores visible before A-layout reload.
        asm volatile("s_wait_dscnt 0" ::: "memory");
        v16bf Aa;
        {
            const __bf16* p = &ldsA[wave][n16][hi * 16];
            v8bf lo = *(const v8bf*)p;
            v8bf hb = *(const v8bf*)(p + 8);
            Aa = __builtin_shufflevector(lo, hb, 0,1,2,3,4,5,6,7,8,9,10,11,12,13,14,15);
        }
        deg = wmma_bf16(Aa, ones, deg);               // rowsum via ones-B
#pragma unroll
        for (int fc = 0; fc < FC; ++fc) {
            v16bf Bh = *(const v16bf*)&stageH[b][fc * 16 + n16][hi * 16];
            acc[fc] = wmma_bf16(Aa, Bh, acc[fc]);
        }

        // Handoff: my DMA done, everyone done reading buffer b.
        if (wave == 0 && j + 32 < N_TOT) __builtin_amdgcn_s_wait_tensorcnt(0);
        __syncthreads();
    }

#pragma unroll
    for (int v = 0; v < 8; ++v) {
        const float rd = __builtin_amdgcn_rcpf(deg[v]);
        const int row = ibase + v + 8 * hi;
#pragma unroll
        for (int fc = 0; fc < FC; ++fc) {
            float val = acc[fc][v] * rd;
            if (RELU) val = fmaxf(val, 0.0f);
            xout[(size_t)row * F + fc * 16 + n16] = val;
        }
    }
}

// ---------------------------------------------------------------------------
// Softmax over 64 columns, one wave per row.
// ---------------------------------------------------------------------------
__global__ void softmax64(const float* __restrict__ x, float* __restrict__ out)
{
    const int wave = threadIdx.x >> 5, lane = threadIdx.x & 31;
    const int row  = blockIdx.x * 8 + wave;
    float a = x[(size_t)row * 64 + lane];
    float b = x[(size_t)row * 64 + lane + 32];
    float m = fmaxf(a, b);
#pragma unroll
    for (int off = 16; off > 0; off >>= 1) m = fmaxf(m, __shfl_xor(m, off, 32));
    float e0 = __expf(a - m), e1 = __expf(b - m);
    float s = e0 + e1;
#pragma unroll
    for (int off = 16; off > 0; off >>= 1) s += __shfl_xor(s, off, 32);
    const float rs = __builtin_amdgcn_rcpf(s);
    out[(size_t)row * 64 + lane]      = e0 * rs;
    out[(size_t)row * 64 + lane + 32] = e1 * rs;
}

// ---------------------------------------------------------------------------
extern "C" void kernel_launch(void* const* d_in, const int* in_sizes, int n_in,
                              void* d_out, int out_size, void* d_ws, size_t ws_size,
                              hipStream_t stream) {
    const float* feat  = (const float*)d_in[0];
    const float* Wgl0  = (const float*)d_in[6];
    const float* bgl0  = (const float*)d_in[7];
    const float* Wgnn0 = (const float*)d_in[8];
    const float* bgnn0 = (const float*)d_in[9];
    const float* Wgl1  = (const float*)d_in[10];
    const float* bgl1  = (const float*)d_in[11];
    const float* Wgnn1 = (const float*)d_in[12];
    const float* bgnn1 = (const float*)d_in[13];
    const float* temp  = (const float*)d_in[14];
    const float* theta = (const float*)d_in[15];
    float* out = (float*)d_out;

    char* ws = (char*)d_ws;
    // all offsets 256B aligned
    __bf16* gl0 = (__bf16*)(ws + 0);                        // 2 MB
    __bf16* hT0 = (__bf16*)(ws + (2u << 20));               // 2 MB
    float*  sq0 = (float*) (ws + (4u << 20));               // 32 KB
    float*  x1  = (float*) (ws + (4u << 20) + (64u << 10)); // 4 MB
    __bf16* gl1 = (__bf16*)(ws + (8u << 20) + (64u << 10)); // 1 MB
    __bf16* hT1 = (__bf16*)(ws + (9u << 20) + (64u << 10)); // 1 MB
    float*  sq1 = (float*) (ws + (10u << 20) + (64u << 10));// 32 KB
    float*  x2  = (float*) (ws + (10u << 20) + (128u << 10));// 2 MB

    // Layer 0
    prep_layer<256, 128><<<N_TOT / 8, 256, 0, stream>>>(feat, Wgl0, bgl0, Wgnn0, bgnn0,
                                                        gl0, hT0, sq0);
    fused_adj_layer<128, true><<<N_TOT / 128, 256, 0, stream>>>(gl0, hT0, sq0, temp, theta, x1);
    // Layer 1
    prep_layer<128, 64><<<N_TOT / 8, 256, 0, stream>>>(x1, Wgl1, bgl1, Wgnn1, bgnn1,
                                                       gl1, hT1, sq1);
    fused_adj_layer<64, false><<<N_TOT / 128, 256, 0, stream>>>(gl1, hT1, sq1, temp, theta, x2);
    // Softmax
    softmax64<<<N_TOT / 8, 256, 0, stream>>>(x2, out);
}
